// UniformMoERouter_38165079392678
// MI455X (gfx1250) — matrix operand order repaired and checked by reading, not compiled, not run on hardware
//
#include <hip/hip_runtime.h>
#include <hip/hip_bf16.h>

// ---------------------------------------------------------------------------
// MoE router for MI455X (gfx1250, wave32, WMMA).
// Gate MLP: one-time fp32->bf16 conversion (x and transposed weights), then
// pure-bf16 tiled GEMMs with v_wmma_f32_16x16x32_bf16 (fp32 accum), bf16
// intermediates. Hot-loop staging uses GLOBAL_LOAD_ASYNC_TO_LDS_B128
// (ASYNCcnt-tracked) when the builtin is available.
// Routing reproduces the serial reference semantics via rank-by-counting +
// one serial redistribution scan.
// ---------------------------------------------------------------------------

#ifndef __has_builtin
#define __has_builtin(x) 0
#endif

#if __has_builtin(__builtin_amdgcn_global_load_async_to_lds_b128)
#define ASYNC_LDS 1
#else
#define ASYNC_LDS 0
#endif

typedef __attribute__((ext_vector_type(16))) __bf16 v16bf;
typedef __attribute__((ext_vector_type(8)))  __bf16 v8bf;
typedef __attribute__((ext_vector_type(8)))  float  v8f;
typedef __attribute__((ext_vector_type(4)))  int    v4i;

#define N_TOK 24576
#define E_EXP 3
#define CAP   8192       // n / E, n % E == 0 -> target == CAP for all experts
#define D_IN  2048

__device__ inline v16bf cat16(v8bf lo, v8bf hi) {
  v16bf r;
#pragma unroll
  for (int i = 0; i < 8; ++i) { r[i] = lo[i]; r[i + 8] = hi[i]; }
  return r;
}

#if ASYNC_LDS
__device__ inline void wait_async0() {
#if __has_builtin(__builtin_amdgcn_s_wait_asynccnt)
  __builtin_amdgcn_s_wait_asynccnt(0);
#else
  asm volatile("s_wait_asynccnt 0x0" ::: "memory");
#endif
}

// builtin signature (from hipcc diagnostic): arg0 = v4i addrspace(1)* (global
// src), arg1 = LDS dst, then i32 offset, i32 cpol.
__device__ inline void async_copy_b128(const void* g, void* l) {
  __builtin_amdgcn_global_load_async_to_lds_b128(
      (v4i __attribute__((address_space(1)))*)(void*)g,
      (v4i __attribute__((address_space(3)))*)l, 0, 0);
}
#endif

// ---------------------------------------------------------------------------
// One-time conversions (outside the GEMM hot loop)
// ---------------------------------------------------------------------------
__global__ void cvt_f32_bf16(const float* __restrict__ in, __bf16* __restrict__ out,
                             long n) {
  long i = ((long)blockIdx.x * blockDim.x + threadIdx.x) * 8;
  if (i >= n) return;
  const float4 a = *(const float4*)(in + i);
  const float4 b = *(const float4*)(in + i + 4);
  v8bf o;
  o[0] = (__bf16)a.x; o[1] = (__bf16)a.y; o[2] = (__bf16)a.z; o[3] = (__bf16)a.w;
  o[4] = (__bf16)b.x; o[5] = (__bf16)b.y; o[6] = (__bf16)b.z; o[7] = (__bf16)b.w;
  *(v8bf*)(out + i) = o;
}

// W [K,N] fp32 -> Wt [N,K] bf16 (writes coalesced along k)
__global__ void cvt_transpose(const float* __restrict__ W, __bf16* __restrict__ Wt,
                              int K, int N) {
  long id = (long)blockIdx.x * blockDim.x + threadIdx.x;
  if (id >= (long)K * N) return;
  int n = (int)(id / K), k = (int)(id % K);
  Wt[(size_t)n * K + k] = (__bf16)W[(size_t)k * N + n];
}

// ---------------------------------------------------------------------------
// Tiled GEMM: C[M,N] = relu(A[M,K] @ B[K,N] + bias), all-bf16 data,
// B given pre-transposed as Bt[N,K]. fp32 accumulate via WMMA 16x16x32.
// Block tile 128x128, K-step 32, 256 threads = 8 waves, wave grid 4x2,
// each wave computes a 32x64 strip = 2x4 WMMA tiles. Double-buffered LDS.
// ---------------------------------------------------------------------------
__global__ __launch_bounds__(256) void gemm_bf16_wmma(
    const __bf16* __restrict__ A,   // [M,K] row-major bf16
    const __bf16* __restrict__ Bt,  // [N,K] row-major bf16 (B transposed)
    const float*  __restrict__ bias,
    __bf16* __restrict__ C,         // [M,N] row-major bf16
    int M, int N, int K)
{
  constexpr int BM = 128, BN = 128, BK = 32;
  constexpr int STR = BK + 8;                 // 40 bf16 -> 80B row pitch (conflict pad)
  __shared__ __bf16 As[2][BM * STR];          // [row][k]
  __shared__ __bf16 Bs[2][BN * STR];          // [col][k]

  const int tid  = threadIdx.x;
  const int lane = tid & 31;
  const int wave = tid >> 5;
  const int wr   = wave >> 1;                 // 0..3 (row waves)
  const int wc   = wave & 1;                  // 0..1 (col waves)
  const int m0   = blockIdx.y * BM;
  const int n0   = blockIdx.x * BN;
  const int lrow = lane & 15;
  const int lhi  = lane >> 4;                 // 0/1 half-wave selector

  v8f zero = {};
  v8f acc[2][4];
#pragma unroll
  for (int i = 0; i < 2; ++i)
#pragma unroll
    for (int j = 0; j < 4; ++j) acc[i][j] = zero;

  const int KT = K / BK;

#if ASYNC_LDS
  // Async global->LDS staging: 512 b128 chunks per 128x32 bf16 tile, 2/thread.
  auto stageA = [&](int kt, int buf) {
#pragma unroll
    for (int it = 0; it < 2; ++it) {
      int idx = tid + it * 256;               // 0..511
      int r = idx >> 2, seg = idx & 3;        // row, 8-bf16 segment
      async_copy_b128(A + (size_t)(m0 + r) * K + kt * BK + seg * 8,
                      &As[buf][r * STR + seg * 8]);
    }
  };
  auto stageB = [&](int kt, int buf) {
#pragma unroll
    for (int it = 0; it < 2; ++it) {
      int idx = tid + it * 256;
      int c = idx >> 2, seg = idx & 3;
      async_copy_b128(Bt + (size_t)(n0 + c) * K + kt * BK + seg * 8,
                      &Bs[buf][c * STR + seg * 8]);
    }
  };

  stageA(0, 0); stageB(0, 0);
  wait_async0();
  __syncthreads();
#else
  v8bf ra[2], rb[2];
  auto loadA = [&](int kt) {
#pragma unroll
    for (int it = 0; it < 2; ++it) {
      int idx = tid + it * 256;
      int r = idx >> 2, seg = idx & 3;
      ra[it] = *(const v8bf*)(A + (size_t)(m0 + r) * K + kt * BK + seg * 8);
    }
  };
  auto loadB = [&](int kt) {
#pragma unroll
    for (int it = 0; it < 2; ++it) {
      int idx = tid + it * 256;
      int c = idx >> 2, seg = idx & 3;
      rb[it] = *(const v8bf*)(Bt + (size_t)(n0 + c) * K + kt * BK + seg * 8);
    }
  };
  auto storeA = [&](int buf) {
#pragma unroll
    for (int it = 0; it < 2; ++it) {
      int idx = tid + it * 256;
      int r = idx >> 2, seg = idx & 3;
      *(v8bf*)&As[buf][r * STR + seg * 8] = ra[it];
    }
  };
  auto storeB = [&](int buf) {
#pragma unroll
    for (int it = 0; it < 2; ++it) {
      int idx = tid + it * 256;
      int c = idx >> 2, seg = idx & 3;
      *(v8bf*)&Bs[buf][c * STR + seg * 8] = rb[it];
    }
  };

  loadA(0); loadB(0); storeA(0); storeB(0);
  __syncthreads();
#endif

  for (int kt = 0; kt < KT; ++kt) {
    const int buf = kt & 1;
#if ASYNC_LDS
    if (kt + 1 < KT) { stageA(kt + 1, buf ^ 1); stageB(kt + 1, buf ^ 1); }
#else
    if (kt + 1 < KT) { loadA(kt + 1); loadB(kt + 1); }
#endif

    // A fragment: lane = row lrow; K = {kb..kb+7, 16+kb..16+kb+7}, kb = lhi*8
    v16bf af[2];
#pragma unroll
    for (int rf = 0; rf < 2; ++rf) {
      const __bf16* p = &As[buf][(wr * 32 + rf * 16 + lrow) * STR + lhi * 8];
      af[rf] = cat16(*(const v8bf*)p, *(const v8bf*)(p + 16));
    }
    // B fragment: lane = col lrow; K = {kb..kb+7, 16+kb..16+kb+7}, kb = lhi*8
    v16bf bfm[4];
#pragma unroll
    for (int cf = 0; cf < 4; ++cf) {
      const __bf16* p = &Bs[buf][(wc * 64 + cf * 16 + lrow) * STR + lhi * 8];
      bfm[cf] = cat16(*(const v8bf*)p, *(const v8bf*)(p + 16));
    }
#pragma unroll
    for (int rf = 0; rf < 2; ++rf)
#pragma unroll
      for (int cf = 0; cf < 4; ++cf)
        acc[rf][cf] = __builtin_amdgcn_wmma_f32_16x16x32_bf16(
            false, af[rf], false, bfm[cf], (short)0, acc[rf][cf], false, false);

#if ASYNC_LDS
    if (kt + 1 < KT) wait_async0();
#else
    if (kt + 1 < KT) { storeA(buf ^ 1); storeB(buf ^ 1); }
#endif
    __syncthreads();
  }

  // Epilogue: C/D layout -> lane col = lrow, VGPR v row = lhi*8 + v
#pragma unroll
  for (int cf = 0; cf < 4; ++cf) {
    const int col = n0 + wc * 64 + cf * 16 + lrow;
    const float bv = bias[col];
#pragma unroll
    for (int rf = 0; rf < 2; ++rf) {
#pragma unroll
      for (int v = 0; v < 8; ++v) {
        const int row = m0 + wr * 32 + rf * 16 + lhi * 8 + v;
        C[(size_t)row * N + col] = (__bf16)fmaxf(acc[rf][cf][v] + bv, 0.0f);
      }
    }
  }
}

// ---------------------------------------------------------------------------
// Last layer (128 -> 3) + softmax + top-1 + stats. One token per thread.
// ---------------------------------------------------------------------------
__global__ void gate4_softmax(const __bf16* __restrict__ H, const float* __restrict__ W4,
                              const float* __restrict__ b4, float* __restrict__ probs,
                              int* __restrict__ choice, float* __restrict__ ptop,
                              float* __restrict__ key1, int* __restrict__ counts,
                              float* __restrict__ probsum, int* __restrict__ assign)
{
  int i = blockIdx.x * blockDim.x + threadIdx.x;
  if (i >= N_TOK) return;
  const __bf16* h = H + (size_t)i * 128;
  float l0 = b4[0], l1 = b4[1], l2 = b4[2];
#pragma unroll 4
  for (int k = 0; k < 128; ++k) {
    float hv = (float)h[k];
    l0 = fmaf(hv, W4[k * 3 + 0], l0);
    l1 = fmaf(hv, W4[k * 3 + 1], l1);
    l2 = fmaf(hv, W4[k * 3 + 2], l2);
  }
  float m = fmaxf(l0, fmaxf(l1, l2));
  float e0 = __expf(l0 - m), e1 = __expf(l1 - m), e2 = __expf(l2 - m);
  float inv = 1.0f / (e0 + e1 + e2);
  float p0 = e0 * inv, p1 = e1 * inv, p2 = e2 * inv;
  probs[i * 3 + 0] = p0; probs[i * 3 + 1] = p1; probs[i * 3 + 2] = p2;
  int best = 0; float pb = p0;
  if (p1 > pb) { best = 1; pb = p1; }
  if (p2 > pb) { best = 2; pb = p2; }
  choice[i] = best; assign[i] = best; ptop[i] = pb;
  key1[i] = (float)best * 2.0f + (1.0f - pb);
  atomicAdd(&counts[best], 1);
  atomicAdd(&probsum[0], p0);
  atomicAdd(&probsum[1], p1);
  atomicAdd(&probsum[2], p2);
}

__global__ void init_small(int* counts, float* probsum, int* countsFinal) {
  int t = threadIdx.x;
  if (t < 3) { counts[t] = 0; probsum[t] = 0.0f; countsFinal[t] = 0; }
}

// rank within expert by (key1 asc, index asc) counting; mark overflow drops
__global__ void rank1_k(const float* __restrict__ key1, const int* __restrict__ choice,
                        const float* __restrict__ ptop, const int* __restrict__ counts,
                        int* __restrict__ dropped, float* __restrict__ key2)
{
  int i = blockIdx.x * blockDim.x + threadIdx.x;
  if (i >= N_TOK) return;
  const float ki = key1[i];
  int pos = 0;
  for (int j = 0; j < N_TOK; ++j) {
    float kj = key1[j];
    pos += (kj < ki) || (kj == ki && j < i);
  }
  const int ch = choice[i];
  int start = 0;
  for (int e = 0; e < ch; ++e) start += counts[e];
  const int d = (pos - start) >= CAP;
  dropped[i] = d;
  key2[i] = d ? ((float)ch * 2.0f + ptop[i]) : 1e9f;
}

// processing order of dropped tokens (key2 asc, index asc) -> scatter order2
__global__ void rank2_k(const float* __restrict__ key2, int* __restrict__ procr,
                        int* __restrict__ order2)
{
  int i = blockIdx.x * blockDim.x + threadIdx.x;
  if (i >= N_TOK) return;
  const float ki = key2[i];
  int cnt = 0;
  for (int j = 0; j < N_TOK; ++j) {
    float kj = key2[j];
    cnt += (kj < ki) || (kj == ki && j < i);
  }
  procr[i] = cnt;
  order2[cnt] = i;
}

// serial redistribution scan (order-dependent by construction -> 1 thread)
__global__ void route_serial(const int* __restrict__ order2, const float* __restrict__ key2,
                             const float* __restrict__ probs, const int* __restrict__ choice,
                             const int* __restrict__ counts, int* __restrict__ assign)
{
  int cnt[3];
  for (int e = 0; e < 3; ++e) { int c = counts[e]; cnt[e] = c < CAP ? c : CAP; }
  for (int k = 0; k < N_TOK; ++k) {
    const int i = order2[k];
    if (key2[i] >= 1e9f) break;                 // dropped tokens sort first
    const float* p = probs + i * 3;
    const int top = choice[i];
    const int ea = (top == 0) ? 1 : 0;          // the two non-top experts, ea < eb
    const int eb = (top == 2) ? 1 : 2;
    int c0, c1;
    if (p[eb] > p[ea]) { c0 = eb; c1 = ea; } else { c0 = ea; c1 = eb; }
    int chosen = (cnt[c0] < CAP) ? c0 : ((cnt[c1] < CAP) ? c1 : c0);
    cnt[chosen] += 1;
    assign[i] = chosen;
  }
}

__global__ void sortkey3_k(const int* __restrict__ assign, const int* __restrict__ dropped,
                           const int* __restrict__ procr, int* __restrict__ sk3,
                           int* __restrict__ countsFinal)
{
  int i = blockIdx.x * blockDim.x + threadIdx.x;
  if (i >= N_TOK) return;
  const int ap = dropped[i] ? (N_TOK + procr[i]) : i;  // unique append key
  sk3[i] = assign[i] * (2 * N_TOK) + ap;
  atomicAdd(&countsFinal[assign[i]], 1);
}

__global__ void rank3_k(const int* __restrict__ sk3, int* __restrict__ order3)
{
  int i = blockIdx.x * blockDim.x + threadIdx.x;
  if (i >= N_TOK) return;
  const int ki = sk3[i];
  int cnt = 0;
  for (int j = 0; j < N_TOK; ++j) cnt += (sk3[j] < ki);  // keys unique
  order3[cnt] = i;
}

__global__ __launch_bounds__(256) void gather_rows(const float* __restrict__ x,
    const int* __restrict__ order3, float* __restrict__ out_data, int* __restrict__ out_idx)
{
  const int r = blockIdx.x;
  const int src = order3[r];
  const float4* s = (const float4*)(x + (size_t)src * D_IN);
  float4* d = (float4*)(out_data + (size_t)r * D_IN);
  for (int c = threadIdx.x; c < D_IN / 4; c += 256) d[c] = s[c];
  if (threadIdx.x == 0) out_idx[r] = src;
}

__global__ void loss_k(const float* __restrict__ probsum, const int* __restrict__ countsFinal,
                       float* __restrict__ out_loss)
{
  float l = 0.0f;
  for (int e = 0; e < 3; ++e)
    l += (probsum[e] / (float)N_TOK) * ((float)countsFinal[e] / (float)N_TOK);
  *out_loss = l * (float)E_EXP * 0.1f;
}

// ---------------------------------------------------------------------------
extern "C" void kernel_launch(void* const* d_in, const int* in_sizes, int n_in,
                              void* d_out, int out_size, void* d_ws, size_t ws_size,
                              hipStream_t stream)
{
  const float* x  = (const float*)d_in[0];
  const float* W1 = (const float*)d_in[1];
  const float* b1 = (const float*)d_in[2];
  const float* W2 = (const float*)d_in[3];
  const float* b2 = (const float*)d_in[4];
  const float* W3 = (const float*)d_in[5];
  const float* b3 = (const float*)d_in[6];
  const float* W4 = (const float*)d_in[7];
  const float* b4 = (const float*)d_in[8];

  // workspace layout (~190 MB)
  __bf16* xb  = (__bf16*)d_ws;                          // 24576 x 2048
  __bf16* Wt1 = xb  + (size_t)N_TOK * 2048;             // 1024 x 2048
  __bf16* Wt2 = Wt1 + (size_t)1024 * 2048;              // 512 x 1024
  __bf16* Wt3 = Wt2 + (size_t)512 * 1024;               // 128 x 512
  __bf16* h1  = Wt3 + (size_t)128 * 512;                // 24576 x 1024
  __bf16* h2  = h1  + (size_t)N_TOK * 1024;             // 24576 x 512
  __bf16* h3  = h2  + (size_t)N_TOK * 512;              // 24576 x 128
  float* probs = (float*)(h3 + (size_t)N_TOK * 128);    // 24576 x 3
  int*   choice  = (int*)(probs + (size_t)N_TOK * 3);
  float* ptop    = (float*)(choice + N_TOK);
  float* key1    = ptop + N_TOK;
  int*   dropped = (int*)(key1 + N_TOK);
  float* key2    = (float*)(dropped + N_TOK);
  int*   procr   = (int*)(key2 + N_TOK);
  int*   order2  = procr + N_TOK;
  int*   assign  = order2 + N_TOK;
  int*   sk3     = assign + N_TOK;
  int*   order3  = sk3 + N_TOK;
  int*   counts      = order3 + N_TOK;                  // 3
  float* probsum     = (float*)(counts + 3);            // 3
  int*   countsFinal = (int*)(probsum + 3);             // 3

  float* out_data = (float*)d_out;                              // 3*8192*2048
  float* out_loss = out_data + (size_t)E_EXP * CAP * D_IN;      // 1
  int*   out_idx  = (int*)(out_loss + 1);                       // 24576 (int32)

  const int nb = (N_TOK + 255) / 256;   // 96

  init_small<<<1, 32, 0, stream>>>(counts, probsum, countsFinal);

  // One-time precision conversion / weight transpose
  cvt_f32_bf16<<<(int)(((long)N_TOK * 2048 / 8 + 255) / 256), 256, 0, stream>>>(
      x, xb, (long)N_TOK * 2048);
  cvt_transpose<<<(int)(((long)2048 * 1024 + 255) / 256), 256, 0, stream>>>(W1, Wt1, 2048, 1024);
  cvt_transpose<<<(int)(((long)1024 * 512  + 255) / 256), 256, 0, stream>>>(W2, Wt2, 1024, 512);
  cvt_transpose<<<(int)(((long)512  * 128  + 255) / 256), 256, 0, stream>>>(W3, Wt3, 512, 128);

  // Gate MLP: all-bf16 WMMA GEMMs, fp32 accumulate, bias+ReLU fused
  gemm_bf16_wmma<<<dim3(1024 / 128, N_TOK / 128), 256, 0, stream>>>(
      xb, Wt1, b1, h1, N_TOK, 1024, 2048);
  gemm_bf16_wmma<<<dim3(512 / 128, N_TOK / 128), 256, 0, stream>>>(
      h1, Wt2, b2, h2, N_TOK, 512, 1024);
  gemm_bf16_wmma<<<dim3(128 / 128, N_TOK / 128), 256, 0, stream>>>(
      h2, Wt3, b3, h3, N_TOK, 128, 512);

  gate4_softmax<<<nb, 256, 0, stream>>>(h3, W4, b4, probs, choice, ptop, key1,
                                        counts, probsum, assign);

  rank1_k<<<nb, 256, 0, stream>>>(key1, choice, ptop, counts, dropped, key2);
  rank2_k<<<nb, 256, 0, stream>>>(key2, procr, order2);
  route_serial<<<1, 1, 0, stream>>>(order2, key2, probs, choice, counts, assign);
  sortkey3_k<<<nb, 256, 0, stream>>>(assign, dropped, procr, sk3, countsFinal);
  rank3_k<<<nb, 256, 0, stream>>>(sk3, order3);

  gather_rows<<<N_TOK, 256, 0, stream>>>(x, order3, out_data, out_idx);
  loss_k<<<1, 1, 0, stream>>>(probsum, countsFinal, out_loss);
}
